// DirectedEdgeEncoder_41077067219075
// MI455X (gfx1250) — matrix operand order repaired and checked by reading, not compiled, not run on hardware
//
#include <hip/hip_runtime.h>

// ---------------------------------------------------------------------------
// DirectedEdgeEncoder: out[e, :] = relu(concat(x[row[e]], edge_attr[e]) @ W^T + b)
//   E = 800k, D_NODE = D_EDGE = 64, K = 128, N_OUT = 128
// Memory-bound (~620 MB min traffic vs 23.3 TB/s => ~27-35 us floor).
//   * bf16 WMMA (v_wmma_f32_16x16x32_bf16), f32 accumulate.
//   * B (weights) bf16, pinned in registers per wave (8 waves x 16-col slice).
//   * Gathered A tile staged once per 16-edge block in LDS (read inputs once).
//   * NT stores for out / NT loads for edge_attr: keep the 12.8 MB x table
//     resident in the 192 MB L2 so gathers never touch HBM.
//   * Uniform full-block fast path: epilogue = 8 immediate-offset b32 stores.
// ---------------------------------------------------------------------------

typedef __attribute__((ext_vector_type(16))) __bf16 v16bf;
typedef __attribute__((ext_vector_type(8)))  __bf16 v8bf;
typedef __attribute__((ext_vector_type(8)))  float  v8f;
typedef __attribute__((ext_vector_type(4)))  float  v4f;

#define D_NODE    64
#define D_EDGE    64
#define K_DIM     128   // D_NODE + D_EDGE
#define N_OUT     128
#define EPB       16    // edges per workgroup iteration (one 16-row WMMA tile)
#define LDS_PITCH 136   // bf16 elems per LDS row: 128 + 8 pad (bank-conflict avoidance)

__device__ __forceinline__ v8bf cvt8(v4f a, v4f b) {
  v8bf r;
  r[0] = (__bf16)a[0]; r[1] = (__bf16)a[1]; r[2] = (__bf16)a[2]; r[3] = (__bf16)a[3];
  r[4] = (__bf16)b[0]; r[5] = (__bf16)b[1]; r[6] = (__bf16)b[2]; r[7] = (__bf16)b[3];
  return r;
}

union V16 { v16bf v; v8bf h[2]; };

__global__ __launch_bounds__(256)
void edge_encoder_wmma(const float* __restrict__ x,
                       const float* __restrict__ ea,
                       const int*   __restrict__ eidx,   // edge_index[0][e] = source row
                       const float* __restrict__ W,      // [N_OUT, K_DIM] row-major
                       const float* __restrict__ bias,   // [N_OUT]
                       float* __restrict__ out,          // [E, N_OUT]
                       int E, int nBlocks)
{
  __shared__ __bf16 Atile[EPB * LDS_PITCH];

  const int  tid  = threadIdx.x;
  const int  lane = tid & 31;
  const int  wave = tid >> 5;        // 0..7 -> output col tile [wave*16, wave*16+16)
  const bool hi   = lane >= 16;
  const int  ln   = lane & 15;

  // ---- B fragments: held in registers for the whole kernel.
  // B (K x N): lane L holds column n = L&15; lanes 0-15 cover K=k0..k0+15,
  // lanes 16-31 cover K=k0+16..k0+31; VGPR v packs K = base+2v, base+2v+1.
  // B[k][n] = W[wave*16 + n][k]  (out = h_in @ W^T).
  const int wRow = wave * 16 + ln;
  v16bf Bfrag[4];
#pragma unroll
  for (int c = 0; c < 4; ++c) {
    const int kb = c * 32 + (hi ? 16 : 0);
    const float* wp = W + (size_t)wRow * K_DIM + kb;
    v4f w0 = *(const v4f*)(wp + 0);
    v4f w1 = *(const v4f*)(wp + 4);
    v4f w2 = *(const v4f*)(wp + 8);
    v4f w3 = *(const v4f*)(wp + 12);
    V16 u; u.h[0] = cvt8(w0, w1); u.h[1] = cvt8(w2, w3);
    Bfrag[c] = u.v;
  }
  const float bv = bias[wave * 16 + ln];

  // staging roles: 16 threads per edge, 8 floats (32 B) each
  const int sEdge = tid >> 4;   // 0..15 local edge
  const int sPart = tid & 15;   // 0..15 -> 8-float segment of the 128-wide row

  // per-lane output offset (constant across loop): row sub-offset + column
  const int outVoff = (hi ? 8 * N_OUT : 0) + wave * 16 + ln;

  for (int blk = blockIdx.x; blk < nBlocks; blk += gridDim.x) {
    const int e0 = blk * EPB;
    const bool full = (e0 + EPB) <= E;

    // ---- stage A tile (gather + concat + f32->bf16) into LDS
    {
      const int e = e0 + sEdge;
      v4f f0 = {}, f1 = {};
      if (e < E) {
        if (sPart < 8) {
          const int r = eidx[e];                         // gather: L2-resident x
          const float* src = x + (size_t)r * D_NODE + sPart * 8;
          f0 = *(const v4f*)(src + 0);
          f1 = *(const v4f*)(src + 4);
        } else {
          const float* src = ea + (size_t)e * D_EDGE + (sPart - 8) * 8;
          f0 = __builtin_nontemporal_load((const v4f*)(src + 0));  // streamed
          f1 = __builtin_nontemporal_load((const v4f*)(src + 4));
        }
      }
      *(v8bf*)&Atile[sEdge * LDS_PITCH + sPart * 8] = cvt8(f0, f1);
    }
    __syncthreads();

    // ---- acc[16x16] = A[16x128] x B[128x16], K in 4 chunks of 32
    // A (16-bit, 16x32): lanes 0-15 row m hold K k0..k0+15;
    // lanes 16-31 hold K k0+16..k0+31.
    v8f acc = {};
#pragma unroll
    for (int c = 0; c < 4; ++c) {
      const int kb = c * 32 + (hi ? 16 : 0);
      const __bf16* ap = &Atile[ln * LDS_PITCH + kb];
      V16 a;
      a.h[0] = *(const v8bf*)(ap);        // K = kb .. kb+7
      a.h[1] = *(const v8bf*)(ap + 8);    // K = kb+8 .. kb+15
      acc = __builtin_amdgcn_wmma_f32_16x16x32_bf16(
              false, a.v, false, Bfrag[c], (short)0, acc, false, false);
    }

    // ---- epilogue: bias + relu, then 8 immediate-offset NT stores
    float res[8];
#pragma unroll
    for (int r = 0; r < 8; ++r) {
      float v = acc[r] + bv;
      res[r] = v > 0.0f ? v : 0.0f;
    }
    float* po = out + (size_t)e0 * N_OUT + outVoff;
    if (full) {
#pragma unroll
      for (int r = 0; r < 8; ++r)
        __builtin_nontemporal_store(res[r], po + r * N_OUT);   // streamed out
    } else {
#pragma unroll
      for (int r = 0; r < 8; ++r)
        if (e0 + r + (hi ? 8 : 0) < E) po[r * N_OUT] = res[r];
    }
    __syncthreads();   // protect LDS before next block's staging
  }
}

extern "C" void kernel_launch(void* const* d_in, const int* in_sizes, int n_in,
                              void* d_out, int out_size, void* d_ws, size_t ws_size,
                              hipStream_t stream) {
  const float* x  = (const float*)d_in[0];
  const float* ea = (const float*)d_in[1];
  const int*   ei = (const int*)  d_in[2];   // edge_index flat [2, E]; use row 0
  const float* W  = (const float*)d_in[3];
  const float* b  = (const float*)d_in[4];
  float*       out = (float*)d_out;

  const int E = in_sizes[1] / D_EDGE;        // edge_attr is [E, 64]
  const int nBlocks = (E + EPB - 1) / EPB;
  int grid = nBlocks < 4096 ? nBlocks : 4096;  // grid-stride: amortize B reg-load
  edge_encoder_wmma<<<grid, 256, 0, stream>>>(x, ea, ei, W, b, out, E, nBlocks);
}
